// Attention_10754598109285
// MI455X (gfx1250) — matrix operand order repaired and checked by reading, not compiled, not run on hardware
//
#include <hip/hip_runtime.h>
#include <hip/hip_bf16.h>

// Problem dims
#define Bn 8
#define Sn 2048
#define Cn 128
#define Hn 4
#define Dn 32
#define Fn (3 * Hn * Dn)   // 384
#define HDn (Hn * Dn)      // 128
#define QK_SCALE 0.17677669529663687f  // 32^-0.5
#define GN_EPS 1e-5f

typedef __attribute__((ext_vector_type(16))) _Float16 v16h;
typedef __attribute__((ext_vector_type(8)))  _Float16 v8h;
typedef __attribute__((ext_vector_type(4)))  _Float16 v4h;
typedef __attribute__((ext_vector_type(8)))  float    v8f;

union V16U { v16h v; v8h h[2]; };
union V4U  { v4h v; _Float16 e[4]; };

// ---------------------------------------------------------------------------
// WMMA helpers (gfx1250 wave32).  16x16x32 f16 -> f32.
// A layout (16x32): lanes 0-15 row M=lane, elems = K{0..7,16..23};
//                   lanes 16-31 row M=lane-16, elems = K{8..15,24..31}.
// B layout (32x16): lanes 0-15 col N=lane,   elems = K{0..7,16..23};
//                   lanes 16-31 col N=lane-16, elems = K{8..15,24..31}.
// C/D layout (16x16 f32): lane n holds col N=(n&15); VGPR g holds row g (lanes
//                         0-15) or row 8+g (lanes 16-31).
// ---------------------------------------------------------------------------
__device__ __forceinline__ v8f wmma_f16(v16h a, v16h b, v8f c) {
    return __builtin_amdgcn_wmma_f32_16x16x32_f16(
        false, a, false, b, (short)0, c, false, false);
}

// Load a 16(M) x 32(K) A-operand tile from row-major [rows x ld] f16 matrix.
__device__ __forceinline__ v16h load_a_tile(const _Float16* base, int row0,
                                            int ld, int k0) {
    int lane = threadIdx.x & 31;
    int half = lane >> 4;
    int l    = lane & 15;
    const _Float16* p = base + (row0 + l) * ld + k0 + half * 8;
    V16U r;
    r.h[0] = *(const v8h*)p;
    r.h[1] = *(const v8h*)(p + 16);
    return r.v;
}

// Load a 32(K) x 16(N) B-operand tile.  `bt` is B-transposed, i.e. row-major
// [N x ldt]:  bt[n*ldt + k] == B[k][n].
__device__ __forceinline__ v16h load_bt_tile(const _Float16* bt, int col0,
                                             int ldt, int k0) {
    int lane = threadIdx.x & 31;
    int half = lane >> 4;
    int l    = lane & 15;
    const _Float16* p = bt + (col0 + l) * ldt + k0 + half * 8;
    V16U r;
    r.h[0] = *(const v8h*)p;
    r.h[1] = *(const v8h*)(p + 16);
    return r.v;
}

// Reductions over each 16-lane half (xor masks < 16 stay within a half).
__device__ __forceinline__ float halfmax(float v) {
    v = fmaxf(v, __shfl_xor(v, 1, 32));
    v = fmaxf(v, __shfl_xor(v, 2, 32));
    v = fmaxf(v, __shfl_xor(v, 4, 32));
    v = fmaxf(v, __shfl_xor(v, 8, 32));
    return v;
}
__device__ __forceinline__ float halfsum(float v) {
    v += __shfl_xor(v, 1, 32);
    v += __shfl_xor(v, 2, 32);
    v += __shfl_xor(v, 4, 32);
    v += __shfl_xor(v, 8, 32);
    return v;
}

// ---------------------------------------------------------------------------
// K1: GroupNorm(1) stats per batch: stats[2b]=mean, stats[2b+1]=rsqrt(var+eps)
// ---------------------------------------------------------------------------
__global__ void gn_stats_k(const float* __restrict__ x, float* __restrict__ stats) {
    int b = blockIdx.x;
    const float4* xb = (const float4*)(x + (size_t)b * Sn * Cn);
    const int n4 = Sn * Cn / 4;
    float s = 0.f, ss = 0.f;
    for (int i = threadIdx.x; i < n4; i += blockDim.x) {
        float4 v = xb[i];
        s  += v.x + v.y + v.z + v.w;
        ss += v.x * v.x + v.y * v.y + v.z * v.z + v.w * v.w;
    }
    __shared__ float as_[256], bs_[256];
    as_[threadIdx.x] = s;
    bs_[threadIdx.x] = ss;
    __syncthreads();
    for (int o = 128; o > 0; o >>= 1) {
        if ((int)threadIdx.x < o) {
            as_[threadIdx.x] += as_[threadIdx.x + o];
            bs_[threadIdx.x] += bs_[threadIdx.x + o];
        }
        __syncthreads();
    }
    if (threadIdx.x == 0) {
        const float inv_n = 1.0f / (float)(Sn * Cn);
        float mu  = as_[0] * inv_n;
        float var = bs_[0] * inv_n - mu * mu;
        stats[2 * b]     = mu;
        stats[2 * b + 1] = rsqrtf(var + GN_EPS);
    }
}

// ---------------------------------------------------------------------------
// K2: normalized x -> f16, row-major [B*S, C]
// ---------------------------------------------------------------------------
__global__ void normalize_k(const float* __restrict__ x,
                            const float* __restrict__ gamma,
                            const float* __restrict__ beta,
                            const float* __restrict__ stats,
                            _Float16* __restrict__ nx) {
    int t  = blockIdx.x * blockDim.x + threadIdx.x;  // 0 .. B*S*C/4-1
    int r  = t >> 5;          // row (b*S+s)
    int c0 = (t & 31) * 4;    // channel quad
    int b  = r >> 11;
    float mu = stats[2 * b];
    float rs = stats[2 * b + 1];
    float4 xv = *(const float4*)(x + (size_t)r * Cn + c0);
    float4 g  = *(const float4*)(gamma + c0);
    float4 be = *(const float4*)(beta + c0);
    V4U o;
    o.e[0] = (_Float16)((xv.x - mu) * rs * g.x + be.x);
    o.e[1] = (_Float16)((xv.y - mu) * rs * g.y + be.y);
    o.e[2] = (_Float16)((xv.z - mu) * rs * g.z + be.z);
    o.e[3] = (_Float16)((xv.w - mu) * rs * g.w + be.w);
    *(v4h*)(nx + (size_t)r * Cn + c0) = o.v;
}

// ---------------------------------------------------------------------------
// K3: convert weights to f16 (layouts already match B^T for WMMA)
// ---------------------------------------------------------------------------
__global__ void wconv_k(const float* __restrict__ wq, const float* __restrict__ wo,
                        _Float16* __restrict__ wqh, _Float16* __restrict__ woh) {
    int t = blockIdx.x * blockDim.x + threadIdx.x;
    const int nq = Fn * Cn;       // 49152
    const int no = Cn * HDn;      // 16384
    if (t < nq) wqh[t] = (_Float16)wq[t];
    else if (t < nq + no) woh[t - nq] = (_Float16)wo[t - nq];
}

// ---------------------------------------------------------------------------
// K4: QKV GEMM.  M=B*S=16384, N=384, K=128.  One wave per 16x16 tile.
// Q scaled by D^-1/2; Q,K stored [B,H,S,D]; V stored transposed [B,H,D,S].
// ---------------------------------------------------------------------------
__global__ void qkv_gemm_k(const _Float16* __restrict__ nx,
                           const _Float16* __restrict__ wqh,
                           _Float16* __restrict__ q, _Float16* __restrict__ k,
                           _Float16* __restrict__ vT) {
    int w    = threadIdx.x >> 5;
    int lane = threadIdx.x & 31;
    int tile = blockIdx.x * 8 + w;          // < 1024*24
    int jt   = tile % (Fn / 16);
    int it   = tile / (Fn / 16);
    int m0   = it * 16;
    int f0   = jt * 16;

    v8f acc = {};
#pragma unroll
    for (int k0 = 0; k0 < Cn; k0 += 32) {
        v16h a  = load_a_tile(nx, m0, Cn, k0);
        v16h bm = load_bt_tile(wqh, f0, Cn, k0);
        acc = wmma_f16(a, bm, acc);
    }

    int half = lane >> 4;
    int l    = lane & 15;
    int f    = f0 + l;                  // output channel (uniform section/head)
    int sec  = f >> 7;                  // 0=q 1=k 2=v
    int h    = (f >> 5) & 3;
    int d    = f & 31;

    if (sec == 0) {
#pragma unroll
        for (int g = 0; g < 8; ++g) {
            int row = m0 + half * 8 + g;
            int b = row >> 11, s = row & (Sn - 1);
            q[(((b * Hn + h) * Sn) + s) * Dn + d] = (_Float16)(acc[g] * QK_SCALE);
        }
    } else if (sec == 1) {
#pragma unroll
        for (int g = 0; g < 8; ++g) {
            int row = m0 + half * 8 + g;
            int b = row >> 11, s = row & (Sn - 1);
            k[(((b * Hn + h) * Sn) + s) * Dn + d] = (_Float16)acc[g];
        }
    } else {
#pragma unroll
        for (int g = 0; g < 8; ++g) {
            int row = m0 + half * 8 + g;
            int b = row >> 11, s = row & (Sn - 1);
            vT[(((b * Hn + h) * Dn) + d) * Sn + s] = (_Float16)acc[g];
        }
    }
}

// ---------------------------------------------------------------------------
// K5: flash attention.  One wave = 16 query rows; 8 waves/block -> 128 rows.
// grid = B*H * (S/128) blocks.  32 keys per inner step: 2 QK WMMAs + online
// softmax + LDS relayout of P + 2 PV WMMAs.
// ---------------------------------------------------------------------------
__global__ void attn_k(const _Float16* __restrict__ q,
                       const _Float16* __restrict__ kk,
                       const _Float16* __restrict__ vT,
                       _Float16* __restrict__ o) {
    __shared__ __align__(16) _Float16 plds[8][16][32];  // per-wave P tile

    int w    = threadIdx.x >> 5;
    int lane = threadIdx.x & 31;
    int half = lane >> 4;
    int l    = lane & 15;

    const int nchunk = Sn / 128;            // 16
    int bid = blockIdx.x;
    int qc  = bid % nchunk;
    int bh  = bid / nchunk;                 // b*H + h
    const _Float16* Qb = q  + (size_t)bh * Sn * Dn;
    const _Float16* Kb = kk + (size_t)bh * Sn * Dn;
    const _Float16* Vt = vT + (size_t)bh * Dn * Sn;

    int i0 = qc * 128 + w * 16;             // query row base for this wave
    v16h qa = load_a_tile(Qb, i0, Dn, 0);   // whole K-dim (D=32) at once

    v8f o0 = {}, o1 = {};
    float rmax[8], rsum[8];
#pragma unroll
    for (int g = 0; g < 8; ++g) { rmax[g] = -1e30f; rsum[g] = 0.f; }

    for (int j0 = 0; j0 < Sn; j0 += 32) {
        v16h kb0 = load_bt_tile(Kb, j0,      Dn, 0);
        v16h kb1 = load_bt_tile(Kb, j0 + 16, Dn, 0);
        v8f z = {};
        v8f s0 = wmma_f16(qa, kb0, z);
        v8f s1 = wmma_f16(qa, kb1, z);

#pragma unroll
        for (int g = 0; g < 8; ++g) {
            float t    = halfmax(fmaxf(s0[g], s1[g]));
            float mnew = fmaxf(rmax[g], t);
            float corr = __expf(rmax[g] - mnew);
            float p0   = __expf(s0[g] - mnew);
            float p1   = __expf(s1[g] - mnew);
            rsum[g] = rsum[g] * corr + halfsum(p0 + p1);
            o0[g] *= corr;
            o1[g] *= corr;
            rmax[g] = mnew;
            int row = g + half * 8;         // C/D-layout row for this lane
            plds[w][row][l]      = (_Float16)p0;
            plds[w][row][16 + l] = (_Float16)p1;
        }

        // Re-read P in A-operand layout (DS ops are in-order within a wave).
        v16h pa  = load_a_tile(&plds[w][0][0], 0, 32, 0);
        v16h vb0 = load_bt_tile(Vt + j0, 0,  Sn, 0);   // d = 0..15
        v16h vb1 = load_bt_tile(Vt + j0, 16, Sn, 0);   // d = 16..31
        o0 = wmma_f16(pa, vb0, o0);
        o1 = wmma_f16(pa, vb1, o1);
    }

    int b = bh >> 2, h = bh & 3;
#pragma unroll
    for (int g = 0; g < 8; ++g) {
        int s = i0 + half * 8 + g;
        float inv = 1.0f / rsum[g];
        _Float16* op = o + ((size_t)(b * Sn + s)) * HDn + h * Dn;
        op[l]      = (_Float16)(o0[g] * inv);
        op[16 + l] = (_Float16)(o1[g] * inv);
    }
}

// ---------------------------------------------------------------------------
// K6: output projection + bias + residual.  M=16384, N=128, K=128.
// ---------------------------------------------------------------------------
__global__ void outproj_k(const _Float16* __restrict__ o,
                          const _Float16* __restrict__ woh,
                          const float* __restrict__ bout,
                          const float* __restrict__ x,
                          float* __restrict__ out) {
    int w    = threadIdx.x >> 5;
    int lane = threadIdx.x & 31;
    int tile = blockIdx.x * 8 + w;          // < 1024*8
    int ct   = tile & 7;
    int it   = tile >> 3;
    int m0   = it * 16;
    int c0   = ct * 16;

    v8f acc = {};
#pragma unroll
    for (int k0 = 0; k0 < HDn; k0 += 32) {
        v16h a  = load_a_tile(o, m0, HDn, k0);
        v16h bm = load_bt_tile(woh, c0, HDn, k0);
        acc = wmma_f16(a, bm, acc);
    }

    int half = lane >> 4;
    int l    = lane & 15;
    int c    = c0 + l;
    float bo = bout[c];
#pragma unroll
    for (int g = 0; g < 8; ++g) {
        int r = m0 + half * 8 + g;
        out[(size_t)r * Cn + c] = acc[g] + bo + x[(size_t)r * Cn + c];
    }
}

// ---------------------------------------------------------------------------
// Host-side launcher
// ---------------------------------------------------------------------------
extern "C" void kernel_launch(void* const* d_in, const int* in_sizes, int n_in,
                              void* d_out, int out_size, void* d_ws, size_t ws_size,
                              hipStream_t stream) {
    const float* x     = (const float*)d_in[0];
    const float* gamma = (const float*)d_in[1];
    const float* beta  = (const float*)d_in[2];
    const float* w_qkv = (const float*)d_in[3];
    const float* w_out = (const float*)d_in[4];
    const float* b_out = (const float*)d_in[5];
    float* out = (float*)d_out;

    char* ws = (char*)d_ws;
    size_t off = 0;
    auto take = [&](size_t bytes) {
        char* p = ws + off;
        off = (off + bytes + 255) & ~(size_t)255;
        return p;
    };
    float*    stats = (float*)take(2 * Bn * sizeof(float));
    _Float16* nx    = (_Float16*)take((size_t)Bn * Sn * Cn * 2);       // 4 MB
    _Float16* wqh   = (_Float16*)take((size_t)Fn * Cn * 2);            // 96 KB
    _Float16* woh   = (_Float16*)take((size_t)Cn * HDn * 2);           // 32 KB
    _Float16* qb    = (_Float16*)take((size_t)Bn * Hn * Sn * Dn * 2);  // 4 MB
    _Float16* kb    = (_Float16*)take((size_t)Bn * Hn * Sn * Dn * 2);  // 4 MB
    _Float16* vT    = (_Float16*)take((size_t)Bn * Hn * Dn * Sn * 2);  // 4 MB
    _Float16* ob    = (_Float16*)take((size_t)Bn * Sn * HDn * 2);      // 4 MB

    gn_stats_k<<<Bn, 256, 0, stream>>>(x, stats);
    normalize_k<<<(Bn * Sn * Cn / 4) / 256, 256, 0, stream>>>(x, gamma, beta,
                                                              stats, nx);
    wconv_k<<<(Fn * Cn + Cn * HDn) / 256, 256, 0, stream>>>(w_qkv, w_out, wqh, woh);
    qkv_gemm_k<<<(Bn * Sn / 16) * (Fn / 16) / 8, 256, 0, stream>>>(nx, wqh, qb,
                                                                   kb, vT);
    attn_k<<<Bn * Hn * (Sn / 128), 256, 0, stream>>>(qb, kb, vT, ob);
    outproj_k<<<(Bn * Sn / 16) * (HDn / 16) / 8, 256, 0, stream>>>(ob, woh, b_out,
                                                                   x, out);
}
